// ModernHopfieldLayer_28509992911130
// MI455X (gfx1250) — matrix operand order repaired and checked by reading, not compiled, not run on hardware
//
#include <hip/hip_runtime.h>

#define DIM   1024
#define MEMN  4096
#define SCALE 0.3125f   // 1/(sqrt(1024)*0.1)

typedef __attribute__((ext_vector_type(16))) __bf16       v16bf;
typedef __attribute__((ext_vector_type(8)))  float        v8f;
typedef __attribute__((ext_vector_type(4)))  unsigned int v4u;

union AFrag { v16bf v; v4u q[2]; };

__device__ __forceinline__ unsigned short f2bf(float x) {
  unsigned int u = __float_as_uint(x);
  u += 0x7FFFu + ((u >> 16) & 1u);          // round-to-nearest-even
  return (unsigned short)(u >> 16);
}

// ---------------------------------------------------------------------------
// Kernel 1: L2-normalize memory rows -> bf16, stored row-major (for GEMM1 B)
// and transposed (for GEMM2 B). One block per memory row.
// ---------------------------------------------------------------------------
__global__ void __launch_bounds__(256)
prep_mem_kernel(const float* __restrict__ mem,
                unsigned short* __restrict__ memn,   // [MEMN][DIM]
                unsigned short* __restrict__ memT) { // [DIM][MEMN]
  const int m = blockIdx.x;
  const int t = threadIdx.x;
  const float* row = mem + (size_t)m * DIM;
  float v0 = row[t], v1 = row[t + 256], v2 = row[t + 512], v3 = row[t + 768];
  float s = v0 * v0 + v1 * v1 + v2 * v2 + v3 * v3;
  // wave32 reduction
  for (int off = 16; off > 0; off >>= 1) s += __shfl_xor(s, off);
  __shared__ float red;
  if (t == 0) red = 0.0f;
  __syncthreads();
  if ((t & 31) == 0) atomicAdd(&red, s);
  __syncthreads();
  const float inv = 1.0f / fmaxf(sqrtf(red), 1e-12f);
  const unsigned short h0 = f2bf(v0 * inv), h1 = f2bf(v1 * inv);
  const unsigned short h2 = f2bf(v2 * inv), h3 = f2bf(v3 * inv);
  const size_t rb = (size_t)m * DIM;
  memn[rb + t] = h0; memn[rb + t + 256] = h1;
  memn[rb + t + 512] = h2; memn[rb + t + 768] = h3;
  memT[(size_t)(t)       * MEMN + m] = h0;
  memT[(size_t)(t + 256) * MEMN + m] = h1;
  memT[(size_t)(t + 512) * MEMN + m] = h2;
  memT[(size_t)(t + 768) * MEMN + m] = h3;
}

// ---------------------------------------------------------------------------
// Kernel 2: fused  softmax(xn @ mn^T * s) @ mn  + residual + LayerNorm.
// Block = 16 batch rows, 256 threads = 8 waves. Streaming over M; output
// accumulators live in registers (8 x v8f per lane).
// ---------------------------------------------------------------------------
__global__ void __launch_bounds__(256)
hopfield_kernel(const float* __restrict__ x,
                const unsigned short* __restrict__ memn,
                const unsigned short* __restrict__ memT,
                const float* __restrict__ gamma,
                const float* __restrict__ beta,
                float* __restrict__ out) {
  __shared__ unsigned short xn[16 * DIM];    // 32KB  x_norm (bf16)
  __shared__ unsigned short pbuf[16 * 256];  //  8KB  exp(sim) tile (bf16)
  __shared__ float rowsq[16];
  __shared__ float denom[16];
  __shared__ float redS[16];
  __shared__ float redQ[16];

  const int tid  = threadIdx.x;
  const int lane = tid & 31;
  const int wave = tid >> 5;
  const int b0   = blockIdx.x * 16;
  const int n    = lane & 15;   // A-row / B-col index
  const int kh   = lane >> 4;   // K-half select

  if (tid < 16) { rowsq[tid] = 0.f; denom[tid] = 0.f; redS[tid] = 0.f; redQ[tid] = 0.f; }
  __syncthreads();

  // ---- normalize x rows into LDS (bf16) ----
  {
    const int r = tid >> 4;          // row 0..15
    const int c = tid & 15;          // 64-elem chunk
    const float* xr = x + (size_t)(b0 + r) * DIM + c * 64;
    float s = 0.f;
    #pragma unroll
    for (int i = 0; i < 64; ++i) { float v = xr[i]; s += v * v; }
    atomicAdd(&rowsq[r], s);
    __syncthreads();
    const float inv = 1.0f / fmaxf(sqrtf(rowsq[r]), 1e-12f);
    unsigned short* dst = &xn[r * DIM + c * 64];
    #pragma unroll
    for (int i = 0; i < 64; ++i) dst[i] = f2bf(xr[i] * inv);
  }
  __syncthreads();

  v8f acc[8];
  #pragma unroll
  for (int t = 0; t < 8; ++t) acc[t] = (v8f){0.f,0.f,0.f,0.f,0.f,0.f,0.f,0.f};
  float dsum[8];
  #pragma unroll
  for (int j = 0; j < 8; ++j) dsum[j] = 0.f;

  for (int m0 = 0; m0 < MEMN; m0 += 256) {
    // ---- Phase 1: sim + exp for this wave's 32 m-columns ----
    {
      const int mbase = m0 + wave * 32;
      v8f s0 = {}; v8f s1 = {};
      const unsigned short* xrow = &xn[n * DIM];
      const unsigned short* mr0  = memn + (size_t)(mbase + n)      * DIM + kh * 16;
      const unsigned short* mr1  = memn + (size_t)(mbase + 16 + n) * DIM + kh * 16;
      if (m0 + 256 < MEMN) {      // hint the next m-chunk's streaming operands
        __builtin_prefetch(mr0 + (size_t)256 * DIM, 0, 1);
        __builtin_prefetch(mr1 + (size_t)256 * DIM, 0, 1);
      }
      #pragma unroll 4
      for (int k0 = 0; k0 < DIM; k0 += 32) {
        AFrag a, bf0, bf1;
        a.q[0]   = *(const v4u*)(xrow + k0 + kh * 16);
        a.q[1]   = *(const v4u*)(xrow + k0 + kh * 16 + 8);
        bf0.q[0] = *(const v4u*)(mr0 + k0);
        bf0.q[1] = *(const v4u*)(mr0 + k0 + 8);
        bf1.q[0] = *(const v4u*)(mr1 + k0);
        bf1.q[1] = *(const v4u*)(mr1 + k0 + 8);
        s0 = __builtin_amdgcn_wmma_f32_16x16x32_bf16(false, a.v, false, bf0.v,
                                                     (short)0, s0, false, false);
        s1 = __builtin_amdgcn_wmma_f32_16x16x32_bf16(false, a.v, false, bf1.v,
                                                     (short)0, s1, false, false);
      }
      // sim in [-0.3125, 0.3125] -> exp is stable, no running max needed
      const int col = wave * 32 + n;
      #pragma unroll
      for (int j = 0; j < 8; ++j) {
        const int br = kh * 8 + j;
        float p0 = __expf(s0[j] * SCALE);
        float p1 = __expf(s1[j] * SCALE);
        dsum[j] += p0 + p1;                 // register-resident denominator
        pbuf[br * 256 + col]      = f2bf(p0);
        pbuf[br * 256 + col + 16] = f2bf(p1);
      }
    }
    __syncthreads();
    // ---- Phase 2: acc += p @ mem_norm for this wave's 128 d-columns ----
    // kk outer: each LDS A-fragment loaded once, feeds 8 independent WMMAs.
    {
      const unsigned short* btw = memT + (size_t)(wave * 128 + n) * MEMN + m0 + kh * 16;
      if (m0 + 256 < MEMN) __builtin_prefetch(btw + 256, 0, 1);
      #pragma unroll
      for (int kk = 0; kk < 256; kk += 32) {
        AFrag a;
        const unsigned short* prow = &pbuf[n * 256 + kk];
        a.q[0] = *(const v4u*)(prow + kh * 16);
        a.q[1] = *(const v4u*)(prow + kh * 16 + 8);
        #pragma unroll
        for (int t = 0; t < 8; ++t) {
          AFrag bf;
          const unsigned short* bt = btw + (size_t)t * 16 * MEMN + kk;
          bf.q[0] = *(const v4u*)(bt);
          bf.q[1] = *(const v4u*)(bt + 8);
          acc[t] = __builtin_amdgcn_wmma_f32_16x16x32_bf16(false, a.v, false, bf.v,
                                                           (short)0, acc[t], false, false);
        }
      }
    }
    __syncthreads();
  }

  // ---- fold register denominators: shuffle over the 16 lanes sharing a row,
  //      then one LDS atomic per row per wave ----
  #pragma unroll
  for (int j = 0; j < 8; ++j) {
    float v = dsum[j];
    v += __shfl_xor(v, 1); v += __shfl_xor(v, 2);
    v += __shfl_xor(v, 4); v += __shfl_xor(v, 8);
    if (n == 0) atomicAdd(&denom[kh * 8 + j], v);
  }
  __syncthreads();

  // ---- Epilogue: h = retrieved/denom + x ; LayerNorm over DIM ----
  float psum[8], psq[8];
  #pragma unroll
  for (int j = 0; j < 8; ++j) { psum[j] = 0.f; psq[j] = 0.f; }
  #pragma unroll
  for (int t = 0; t < 8; ++t) {
    const int d = wave * 128 + t * 16 + n;
    #pragma unroll
    for (int j = 0; j < 8; ++j) {
      const int br = kh * 8 + j;
      float hv = acc[t][j] / denom[br] + x[(size_t)(b0 + br) * DIM + d];
      acc[t][j] = hv;
      psum[j] += hv;
      psq[j]  += hv * hv;
    }
  }
  #pragma unroll
  for (int j = 0; j < 8; ++j) {
    float s = psum[j], q = psq[j];
    s += __shfl_xor(s, 1); s += __shfl_xor(s, 2);
    s += __shfl_xor(s, 4); s += __shfl_xor(s, 8);
    q += __shfl_xor(q, 1); q += __shfl_xor(q, 2);
    q += __shfl_xor(q, 4); q += __shfl_xor(q, 8);
    if (n == 0) {
      atomicAdd(&redS[kh * 8 + j], s);
      atomicAdd(&redQ[kh * 8 + j], q);
    }
  }
  __syncthreads();
  #pragma unroll
  for (int t = 0; t < 8; ++t) {
    const int d = wave * 128 + t * 16 + n;
    const float g = gamma[d], bb = beta[d];
    #pragma unroll
    for (int j = 0; j < 8; ++j) {
      const int br  = kh * 8 + j;
      const float mu  = redS[br] * (1.0f / DIM);
      const float var = redQ[br] * (1.0f / DIM) - mu * mu;
      const float rst = rsqrtf(var + 1e-5f);
      out[(size_t)(b0 + br) * DIM + d] = (acc[t][j] - mu) * rst * g + bb;
    }
  }
}

// ---------------------------------------------------------------------------
extern "C" void kernel_launch(void* const* d_in, const int* in_sizes, int n_in,
                              void* d_out, int out_size, void* d_ws, size_t ws_size,
                              hipStream_t stream) {
  const float* x     = (const float*)d_in[0];   // (16384, 1024)
  const float* mem   = (const float*)d_in[1];   // (4096, 1024)
  const float* gamma = (const float*)d_in[2];   // (1024,)
  const float* beta  = (const float*)d_in[3];   // (1024,)
  float* out = (float*)d_out;                   // (16384, 1024) f32

  unsigned short* memn = (unsigned short*)d_ws;                 // 8 MB
  unsigned short* memT = memn + (size_t)MEMN * DIM;             // 8 MB

  prep_mem_kernel<<<MEMN, 256, 0, stream>>>(mem, memn, memT);
  hopfield_kernel<<<16384 / 16, 256, 0, stream>>>(x, memn, memT, gamma, beta, out);
}
